// MultiRL_21947282883246
// MI455X (gfx1250) — compile-verified, matrix operand order checked by reading, not run on hardware
//
#include <hip/hip_runtime.h>
#include <hip/hip_bf16.h>
#include <stdint.h>

#define NROWS 2048
#define DIM_IM 1024
#define DIM_CLI 64

typedef __attribute__((ext_vector_type(16))) __bf16        v16bf;
typedef __attribute__((ext_vector_type(8)))  float         v8f;
typedef __attribute__((ext_vector_type(4)))  unsigned int  v4u;
typedef __attribute__((ext_vector_type(4)))  int           v4i;

typedef __attribute__((address_space(1))) v4i* as1_v4i_p;
typedef __attribute__((address_space(3))) v4i* as3_v4i_p;

#if __has_builtin(__builtin_amdgcn_global_load_async_to_lds_b128)
#define USE_ASYNC_LDS 1
#else
#define USE_ASYNC_LDS 0
#endif

union FragU { v4u q[2]; v16bf v; };

// 16-byte global -> LDS copy. Prefer the CDNA5 async-to-LDS path (ASYNCcnt);
// fall back to a plain load/store pair if the builtin is unavailable.
// Builtin signature (from hipcc diagnostic): (v4i addrspace(1)*, v4i addrspace(3)*, imm, imm)
__device__ __forceinline__ void g2s_b128(unsigned short* lds, const unsigned short* g) {
#if USE_ASYNC_LDS
  __builtin_amdgcn_global_load_async_to_lds_b128(
      (as1_v4i_p)(unsigned long long)g,
      (as3_v4i_p)(unsigned int)(unsigned long long)lds,
      /*offset=*/0, /*cpol=*/0);
#else
  *(v4u*)lds = *(const v4u*)g;
#endif
}

__device__ __forceinline__ void wait_async_then_barrier() {
#if USE_ASYNC_LDS
#if __has_builtin(__builtin_amdgcn_s_wait_asynccnt)
  __builtin_amdgcn_s_wait_asynccnt(0);
#else
  asm volatile("s_wait_asynccnt 0" ::: "memory");
#endif
#endif
  __syncthreads();
}

// Per-lane A/B fragment for v_wmma_f32_16x16x32_bf16 from global (row-major, ld=ldk).
// ISA 16-bit A layout: lane (r,h) holds K = [k0, k0+8) in VGPR0..3 and
// K = [k0+16, k0+24) in VGPR4..7, with k0 = kbase + 8*h.
__device__ __forceinline__ v16bf load_frag(const unsigned short* __restrict__ base,
                                           int row, int ldk, int k0) {
  const unsigned short* p = base + (size_t)row * ldk + k0;
  FragU f;
  f.q[0] = *(const v4u*)(p);
  f.q[1] = *(const v4u*)(p + 16);
  return f.v;
}

// Same fragment sourced from LDS (padded row stride) -> ds_load_b128.
__device__ __forceinline__ v16bf frag_lds(const unsigned short* lds, int row, int ldk, int k0) {
  const unsigned short* p = lds + row * ldk + k0;
  FragU f;
  f.q[0] = *(const v4u*)(p);
  f.q[1] = *(const v4u*)(p + 16);
  return f.v;
}

// Monotonic float -> uint mapping (min float == min key)
__device__ __forceinline__ unsigned int fkey(float f) {
  unsigned int u = __float_as_uint(f);
  return (u & 0x80000000u) ? ~u : (u | 0x80000000u);
}

// Row category: 0 = (t1==0 && traumatic), 1 = t1==1, 2 = t1==2, 3 = none
__global__ void prep_cat(const int* __restrict__ t, const int* __restrict__ trauma,
                         int* __restrict__ cat) {
  int i = blockIdx.x * blockDim.x + threadIdx.x;
  if (i < NROWS) {
    int t1 = t[2 * i + 1];
    int c = 3;
    if (t1 == 1) c = 1;
    else if (t1 == 2) c = 2;
    else if (t1 == 0 && trauma[i] == 1) c = 0;
    cat[i] = c;
  }
}

// Per-row L2 normalize, then split each f32 into bf16 hi (truncation, exact
// residual) + bf16 lo (RNE) for the bf16x3 WMMA scheme.
__global__ void prep_row(const float* __restrict__ X,
                         unsigned short* __restrict__ Hi,
                         unsigned short* __restrict__ Lo, int D) {
  __shared__ float red[256];
  int row = blockIdx.x;
  const float* x = X + (size_t)row * D;
  float ss = 0.0f;
  for (int i = threadIdx.x; i < D; i += blockDim.x) { float v = x[i]; ss += v * v; }
  red[threadIdx.x] = ss;
  __syncthreads();
  for (int s = blockDim.x >> 1; s > 0; s >>= 1) {
    if ((int)threadIdx.x < s) red[threadIdx.x] += red[threadIdx.x + s];
    __syncthreads();
  }
  float scale = 1.0f / fmaxf(sqrtf(red[0]), 1e-12f);
  for (int i = threadIdx.x; i < D; i += blockDim.x) {
    float v = x[i] * scale;
    unsigned int u  = __float_as_uint(v);
    unsigned int hb = u & 0xFFFF0000u;            // truncate -> hi exactly representable
    float lf = v - __uint_as_float(hb);           // exact residual in f32
    unsigned int lu = __float_as_uint(lf);
    unsigned short lb = (unsigned short)((lu + 0x7FFFu + ((lu >> 16) & 1u)) >> 16); // RNE
    Hi[(size_t)row * D + i] = (unsigned short)(hb >> 16);
    Lo[(size_t)row * D + i] = lb;
  }
}

__global__ void init_slots(unsigned long long* __restrict__ slots) {
  if (threadIdx.x < 6) slots[threadIdx.x] = ~0ull;
}

// Epilogue for one 16x16 tile: write 3 masked matrices, fold packed argmin keys.
__device__ __forceinline__ void emit_tile(const v8f& acc, int tm, int tn, int r, int h,
                                          const int* __restrict__ cat,
                                          float* __restrict__ outBase,
                                          unsigned long long& K0,
                                          unsigned long long& K1,
                                          unsigned long long& K2) {
  const int Nn = NROWS;
  const size_t NN = (size_t)Nn * Nn;
  int n = tn * 16 + r;
  int cn = cat[n];
#pragma unroll
  for (int d = 0; d < 8; ++d) {
    int m = tm * 16 + d + 8 * h;       // C/D layout: VGPR d -> M = d + 8h, N = r
    int cm = cat[m];
    float s = acc[d];
    unsigned int flat = (unsigned int)(m * Nn + n);
    unsigned long long key = ((unsigned long long)fkey(s) << 32) | (unsigned long long)flat;
    size_t off = (size_t)m * Nn + n;
    bool v0 = (cm == 0) && (cn == 1);
    bool v1 = (cm == 0) && (cn == 2);
    bool v2 = (cm == 1) && (cn == 2);
    outBase[off]          = v0 ? s : 0.0f;
    outBase[NN + off]     = v1 ? s : 0.0f;
    outBase[2 * NN + off] = v2 ? s : 0.0f;
    if (v0 && key < K0) K0 = key;
    if (v1 && key < K1) K1 = key;
    if (v2 && key < K2) K2 = key;
  }
}

// 8 waves/block share tileM. The block stages the 16xK A strip (hi+lo) in LDS
// once (async-to-LDS); each wave computes TWO adjacent 16x16 N-tiles so the
// LDS A-fragments feed two independent WMMA chains per K-chunk.
// acc += hi*hi + hi*lo + lo*hi  (bf16x3 ~ f32 accuracy)
template <int K>
__global__ void gemm_masked(const unsigned short* __restrict__ Xhi,
                            const unsigned short* __restrict__ Xlo,
                            const int* __restrict__ cat,
                            float* __restrict__ outBase,            // 3 masked matrices
                            unsigned long long* __restrict__ slots) { // 3 argmin slots
  constexpr int LK = K + 8;  // +16B row pad: bank-conflict-free ds_load_b128
  __shared__ __align__(16) unsigned short lAhi[16 * LK];
  __shared__ __align__(16) unsigned short lAlo[16 * LK];

  int tm = blockIdx.x;                 // 0..127
  const unsigned short* gAhi = Xhi + (size_t)tm * 16 * K;
  const unsigned short* gAlo = Xlo + (size_t)tm * 16 * K;
  for (int idx = threadIdx.x; idx < 16 * (K / 8); idx += blockDim.x) {
    int row = idx / (K / 8);
    int col = (idx % (K / 8)) * 8;
    g2s_b128(&lAhi[row * LK + col], gAhi + row * K + col);
    g2s_b128(&lAlo[row * LK + col], gAlo + row * K + col);
  }
  wait_async_then_barrier();

  int lane = threadIdx.x & 31;
  int wave = threadIdx.x >> 5;
  int r = lane & 15, h = lane >> 4;
  int tn0 = (blockIdx.y * 8 + wave) * 2;  // 0,2,..,126
  int tn1 = tn0 + 1;
  int brow0 = tn0 * 16 + r;
  int brow1 = tn1 * 16 + r;

  v8f acc0 = {0.f, 0.f, 0.f, 0.f, 0.f, 0.f, 0.f, 0.f};
  v8f acc1 = {0.f, 0.f, 0.f, 0.f, 0.f, 0.f, 0.f, 0.f};
#pragma unroll 2
  for (int kb = 0; kb < K; kb += 32) {
    int k0 = kb + 8 * h;
    v16bf ahi = frag_lds(lAhi, r, LK, k0);
    v16bf alo = frag_lds(lAlo, r, LK, k0);
    v16bf b0h = load_frag(Xhi, brow0, K, k0);
    v16bf b0l = load_frag(Xlo, brow0, K, k0);
    v16bf b1h = load_frag(Xhi, brow1, K, k0);
    v16bf b1l = load_frag(Xlo, brow1, K, k0);
    acc0 = __builtin_amdgcn_wmma_f32_16x16x32_bf16(false, ahi, false, b0h, (short)0, acc0, false, false);
    acc1 = __builtin_amdgcn_wmma_f32_16x16x32_bf16(false, ahi, false, b1h, (short)0, acc1, false, false);
    acc0 = __builtin_amdgcn_wmma_f32_16x16x32_bf16(false, ahi, false, b0l, (short)0, acc0, false, false);
    acc1 = __builtin_amdgcn_wmma_f32_16x16x32_bf16(false, ahi, false, b1l, (short)0, acc1, false, false);
    acc0 = __builtin_amdgcn_wmma_f32_16x16x32_bf16(false, alo, false, b0h, (short)0, acc0, false, false);
    acc1 = __builtin_amdgcn_wmma_f32_16x16x32_bf16(false, alo, false, b1h, (short)0, acc1, false, false);
  }

  unsigned long long K0 = ~0ull, K1 = ~0ull, K2 = ~0ull;
  emit_tile(acc0, tm, tn0, r, h, cat, outBase, K0, K1, K2);
  emit_tile(acc1, tm, tn1, r, h, cat, outBase, K0, K1, K2);
  if (K0 != ~0ull) atomicMin(&slots[0], K0);
  if (K1 != ~0ull) atomicMin(&slots[1], K1);
  if (K2 != ~0ull) atomicMin(&slots[2], K2);
}

__global__ void finalize(const unsigned long long* __restrict__ slots,
                         float* __restrict__ out) {
  int p = threadIdx.x;
  if (p < 6) {
    unsigned long long s = slots[p];
    float v;
    unsigned int i = 0, j = 0;
    if (s == ~0ull) {
      v = __builtin_inff();            // empty mask: argmin of all-inf -> idx 0
    } else {
      unsigned int key = (unsigned int)(s >> 32);
      unsigned int u = (key & 0x80000000u) ? (key ^ 0x80000000u) : ~key;
      v = __uint_as_float(u);
      unsigned int flat = (unsigned int)(s & 0xFFFFFFFFu);
      i = flat / NROWS;
      j = flat % NROWS;
    }
    size_t base = (size_t)6 * NROWS * NROWS;
    out[base + p] = v;
    out[base + 6 + 2 * p + 0] = (float)i;
    out[base + 6 + 2 * p + 1] = (float)j;
  }
}

extern "C" void kernel_launch(void* const* d_in, const int* in_sizes, int n_in,
                              void* d_out, int out_size, void* d_ws, size_t ws_size,
                              hipStream_t stream) {
  const float* phi_im  = (const float*)d_in[0];
  const float* phi_cli = (const float*)d_in[1];
  const int*   t       = (const int*)d_in[2];
  const int*   trauma  = (const int*)d_in[3];
  float* out = (float*)d_out;

  // Workspace layout (bytes):
  char* ws = (char*)d_ws;
  unsigned short* imHi  = (unsigned short*)(ws + 0);         // 2048*1024*2 = 4 MiB
  unsigned short* imLo  = (unsigned short*)(ws + 4194304);   // 4 MiB
  unsigned short* cliHi = (unsigned short*)(ws + 8388608);   // 256 KiB
  unsigned short* cliLo = (unsigned short*)(ws + 8650752);   // 256 KiB
  int*            cat   = (int*)(ws + 8912896);              // 8 KiB
  unsigned long long* slots = (unsigned long long*)(ws + 8921088); // 48 B

  prep_cat<<<dim3(8), dim3(256), 0, stream>>>(t, trauma, cat);
  prep_row<<<dim3(NROWS), dim3(256), 0, stream>>>(phi_im, imHi, imLo, DIM_IM);
  prep_row<<<dim3(NROWS), dim3(64), 0, stream>>>(phi_cli, cliHi, cliLo, DIM_CLI);
  init_slots<<<dim3(1), dim3(32), 0, stream>>>(slots);

  gemm_masked<DIM_IM><<<dim3(128, 8), dim3(256), 0, stream>>>(
      imHi, imLo, cat, out, slots);
  gemm_masked<DIM_CLI><<<dim3(128, 8), dim3(256), 0, stream>>>(
      cliHi, cliLo, cat, out + (size_t)3 * NROWS * NROWS, slots + 3);

  finalize<<<dim3(1), dim3(32), 0, stream>>>(slots, out);
}
